// PDNConvNN_32693291057508
// MI455X (gfx1250) — matrix operand order repaired and checked by reading, not compiled, not run on hardware
//
#include <hip/hip_runtime.h>
#include <hip/hip_bf16.h>
#include <math.h>

// ---------------- problem constants (match reference) ----------------
#define N_NODES   100000
#define N_EDGES   1600000
#define HID       128
#define EDGE_DIM  8
#define HC        32          // edge-MLP hidden
#define IN_FLAT   36          // 6*6 flattened node input
#define ST_DIM    138         // HID + 4 + 6
#define OUT_F     30          // 6*5 decoder output
#define N_LAYERS  5
#define LN_EPS    1e-5f

typedef __attribute__((ext_vector_type(16))) _Float16 v16h;
typedef __attribute__((ext_vector_type(8)))  float    v8f;

// ---------------------------------------------------------------------
// Native CDNA5 f32 add atomic (no return), device scope.
// Guarantees global_atomic_add_f32 (never a CAS loop).
// ---------------------------------------------------------------------
__device__ __forceinline__ void atomic_add_f32_dev(float* p, float v)
{
    asm volatile("global_atomic_add_f32 %0, %1, off scope:SCOPE_DEV"
                 :: "v"((unsigned long long)(uintptr_t)p), "v"(v)
                 : "memory");
}

// ---------------------------------------------------------------------
// Encoder: h = relu(LayerNorm(concat(x@W_emb+b, t, s) @ W_st + b_st))
// one 128-thread block per node
// ---------------------------------------------------------------------
__global__ void k_encode(const float* __restrict__ x,
                         const float* __restrict__ t,
                         const float* __restrict__ s,
                         const float* __restrict__ W_emb, const float* __restrict__ b_emb,
                         const float* __restrict__ W_st,  const float* __restrict__ b_st,
                         const float* __restrict__ ln_g,  const float* __restrict__ ln_b,
                         float* __restrict__ h)
{
    __shared__ float xv[IN_FLAT];
    __shared__ float cvec[ST_DIM];
    __shared__ float red[HID];
    const int node = blockIdx.x;
    const int j    = threadIdx.x;          // 0..127

    if (j < IN_FLAT) xv[j] = x[(size_t)node * IN_FLAT + j];
    __syncthreads();

    float e = b_emb[j];
    #pragma unroll 4
    for (int k = 0; k < IN_FLAT; ++k) e += xv[k] * W_emb[k * HID + j];
    cvec[j] = e;
    if (j < 4) cvec[HID + j]     = t[(size_t)node * 4 + j];
    if (j < 6) cvec[HID + 4 + j] = s[(size_t)node * 6 + j];
    __syncthreads();

    float hv = b_st[j];
    #pragma unroll 4
    for (int k = 0; k < ST_DIM; ++k) hv += cvec[k] * W_st[k * HID + j];

    // LayerNorm over the 128 features via LDS tree reduction
    red[j] = hv; __syncthreads();
    for (int off = 64; off > 0; off >>= 1) { if (j < off) red[j] += red[j + off]; __syncthreads(); }
    const float mu = red[0] * (1.0f / HID);
    __syncthreads();
    const float d = hv - mu;
    red[j] = d * d; __syncthreads();
    for (int off = 64; off > 0; off >>= 1) { if (j < off) red[j] += red[j + off]; __syncthreads(); }
    const float var = red[0] * (1.0f / HID);

    const float y = d * rsqrtf(var + LN_EPS) * ln_g[j] + ln_b[j];
    h[(size_t)node * HID + j] = fmaxf(y, 0.0f);
}

// ---------------------------------------------------------------------
// deg[n] = v  (self-loop weight)
// ---------------------------------------------------------------------
__global__ void k_fill(float* __restrict__ p, float v, int n)
{
    int i = blockIdx.x * blockDim.x + threadIdx.x;
    if (i < n) p[i] = v;
}

// ---------------------------------------------------------------------
// Per-edge weight MLP: w = sigmoid(relu(ea@W1+b1)@W2+b2); deg[col] += w
// ---------------------------------------------------------------------
__global__ void k_edge_w(const float* __restrict__ ea,
                         const int*   __restrict__ col,
                         const float* __restrict__ W1, const float* __restrict__ b1,
                         const float* __restrict__ W2, const float* __restrict__ b2,
                         float* __restrict__ w_e, float* __restrict__ deg)
{
    int e = blockIdx.x * blockDim.x + threadIdx.x;
    if (e >= N_EDGES) return;
    float a[EDGE_DIM];
    #pragma unroll
    for (int k = 0; k < EDGE_DIM; ++k) a[k] = ea[(size_t)e * EDGE_DIM + k];
    float acc = b2[0];
    #pragma unroll 4
    for (int c = 0; c < HC; ++c) {
        float hc = b1[c];
        #pragma unroll
        for (int k = 0; k < EDGE_DIM; ++k) hc += a[k] * W1[k * HC + c];
        acc += fmaxf(hc, 0.0f) * W2[c];
    }
    const float w = 1.0f / (1.0f + __expf(-acc));
    w_e[e] = w;
    atomic_add_f32_dev(&deg[col[e]], w);
}

// ---------------------------------------------------------------------
// dinv = deg > 0 ? rsqrt(deg) : 0
// ---------------------------------------------------------------------
__global__ void k_dinv(const float* __restrict__ deg, float* __restrict__ dinv, int n)
{
    int i = blockIdx.x * blockDim.x + threadIdx.x;
    if (i < n) { float d = deg[i]; dinv[i] = d > 0.0f ? rsqrtf(d) : 0.0f; }
}

// ---------------------------------------------------------------------
// Convert this layer's 128x128 f32 weight to f16 (row-major)
// ---------------------------------------------------------------------
__global__ void k_wcvt(const float* __restrict__ W, _Float16* __restrict__ W16)
{
    int i = blockIdx.x * blockDim.x + threadIdx.x;
    if (i < HID * HID) W16[i] = (_Float16)W[i];
}

// ---------------------------------------------------------------------
// xl = relu(h) @ W   via V_WMMA_F32_16X16X32_F16
// block = 256 threads (8 waves); block handles 16 nodes x 128 outputs;
// wave j owns output columns [16j, 16j+16). K = 128 -> 4 WMMA steps.
//
// Staging: GLOBAL_LOAD_ASYNC_TO_LDS_B128 DMAs the raw f32 tile into LDS
// (ASYNCcnt), then ONE cooperative pass does relu + f16 convert (once per
// element, not per wave), so the WMMA loop is pure ds_load_b128 + wmma.
//
// A-frag (16x32 f16): lane l holds row M=l&15, K = 8*(l>>4)+{0..7} and 16+8*(l>>4)+{0..7}
// B-frag (32x16 f16): lane l holds col N=l&15, K = 8*(l>>4)+{0..7} and 16+8*(l>>4)+{0..7}
// D (16x16 f32):      lane l col N=l&15; VGPR v -> row M = v + 8*(l>>4)
// ---------------------------------------------------------------------
__global__ void __launch_bounds__(256)
k_gemm_wmma(const float* __restrict__ h, const _Float16* __restrict__ W16,
            float* __restrict__ xl)
{
    __shared__ __align__(16) float    stage[16 * HID];   // 8 KB raw f32 tile
    __shared__ __align__(32) _Float16 tileA[16 * HID];   // 4 KB relu(h) as f16

    const int m0  = blockIdx.x * 16;
    const int tid = threadIdx.x;

    // async DMA: 2048 floats = 256 threads x 2 x b128
    {
        const int i0 = tid * 4;            // float index of first b128
        const int i1 = i0 + 1024;          // second half of the tile
        const float* g0 = h + (size_t)m0 * HID + i0;
        const float* g1 = h + (size_t)m0 * HID + i1;
        // low 32 bits of a generic shared pointer = wave-relative LDS byte offset
        const unsigned l0 = (unsigned)(uintptr_t)(&stage[i0]);
        const unsigned l1 = (unsigned)(uintptr_t)(&stage[i1]);
        asm volatile("global_load_async_to_lds_b128 %0, %1, off"
                     :: "v"(l0), "v"((unsigned long long)(uintptr_t)g0) : "memory");
        asm volatile("global_load_async_to_lds_b128 %0, %1, off"
                     :: "v"(l1), "v"((unsigned long long)(uintptr_t)g1) : "memory");
        asm volatile("s_wait_asynccnt 0x0" ::: "memory");
    }
    __syncthreads();

    // one-pass relu + f16 convert (8 contiguous elements per thread)
    {
        const int base = tid * 8;
        #pragma unroll
        for (int q = 0; q < 8; ++q)
            tileA[base + q] = (_Float16)fmaxf(stage[base + q], 0.0f);
    }
    __syncthreads();

    const int wave = tid >> 5;      // N-tile index 0..7
    const int lane = tid & 31;
    const int lm   = lane & 15;
    const int lh   = lane >> 4;     // 0 or 1

    v8f c = {};
    #pragma unroll
    for (int ks = 0; ks < 4; ++ks) {
        const int k0 = ks * 32;
        // A fragment from LDS f16 (two contiguous 8-half runs -> ds_load_b128 pair)
        const _Float16* ap = tileA + lm * HID + k0 + 8 * lh;
        v16h a;
        #pragma unroll
        for (int q = 0; q < 8; ++q) { a[q] = ap[q]; a[8 + q] = ap[16 + q]; }
        // B fragment: column wave*16+lm, K in same order as A (strided gather, L2-hot)
        const _Float16* bp = W16 + (size_t)(k0 + 8 * lh) * HID + wave * 16 + lm;
        v16h b;
        #pragma unroll
        for (int q = 0; q < 8; ++q) { b[q] = bp[q * HID]; b[8 + q] = bp[(16 + q) * HID]; }

        c = __builtin_amdgcn_wmma_f32_16x16x32_f16(
                /*neg_a=*/false, a, /*neg_b=*/false, b,
                /*c_mod=*/(short)0, c, /*reuse_a=*/false, /*reuse_b=*/false);
    }

    // store D
    const int colN = wave * 16 + lm;
    #pragma unroll
    for (int v = 0; v < 8; ++v) {
        const int node = m0 + v + 8 * lh;
        xl[(size_t)node * HID + colN] = c[v];
    }
}

// ---------------------------------------------------------------------
// out[n][k] = conv_b[k] + dinv[n]^2 * xl[n][k]   (self-loop message)
// ---------------------------------------------------------------------
__global__ void k_out_init(const float* __restrict__ xl, const float* __restrict__ dinv,
                           const float* __restrict__ bias, float* __restrict__ out)
{
    int i = blockIdx.x * blockDim.x + threadIdx.x;
    if (i >= N_NODES * HID) return;
    const int n = i >> 7, k = i & (HID - 1);
    const float di = dinv[n];
    out[i] = bias[k] + di * di * xl[i];
}

// ---------------------------------------------------------------------
// Scatter: one wave per edge; 32 lanes x float4 = 128 features.
// out[col] += dinv[row]*w*dinv[col] * xl[row]
// (xl and out are both L2-resident: ~51 MB each vs 192 MB L2)
// ---------------------------------------------------------------------
__global__ void k_scatter(const float* __restrict__ xl,
                          const int* __restrict__ row, const int* __restrict__ col,
                          const float* __restrict__ w_e, const float* __restrict__ dinv,
                          float* __restrict__ out)
{
    const int gw   = (blockIdx.x * blockDim.x + threadIdx.x) >> 5;  // edge id
    const int lane = threadIdx.x & 31;
    if (gw >= N_EDGES) return;
    const int r = row[gw];
    const int c = col[gw];
    const float nrm = dinv[r] * w_e[gw] * dinv[c];
    const float4 v = ((const float4*)(xl + (size_t)r * HID))[lane];
    float* dst = out + (size_t)c * HID + lane * 4;
    atomic_add_f32_dev(dst + 0, nrm * v.x);
    atomic_add_f32_dev(dst + 1, nrm * v.y);
    atomic_add_f32_dev(dst + 2, nrm * v.z);
    atomic_add_f32_dev(dst + 3, nrm * v.w);
}

// ---------------------------------------------------------------------
// Decoder: out = relu(h) @ W_dec + b_dec  (128 -> 30)
// ---------------------------------------------------------------------
__global__ void k_dec(const float* __restrict__ h, const float* __restrict__ Wd,
                      const float* __restrict__ bd, float* __restrict__ out)
{
    const int tgl  = blockIdx.x * blockDim.x + threadIdx.x;
    const int node = tgl >> 5;
    const int j    = tgl & 31;
    if (node >= N_NODES || j >= OUT_F) return;
    float acc = bd[j];
    #pragma unroll 4
    for (int k = 0; k < HID; ++k) {
        acc += fmaxf(h[(size_t)node * HID + k], 0.0f) * Wd[k * OUT_F + j];
    }
    out[(size_t)node * OUT_F + j] = acc;
}

// ---------------------------------------------------------------------
extern "C" void kernel_launch(void* const* d_in, const int* in_sizes, int n_in,
                              void* d_out, int out_size, void* d_ws, size_t ws_size,
                              hipStream_t stream)
{
    const float* x      = (const float*)d_in[0];
    const int*   eidx   = (const int*)  d_in[1];
    const float* ea     = (const float*)d_in[2];
    const float* t      = (const float*)d_in[3];
    const float* s      = (const float*)d_in[4];
    const float* W_emb  = (const float*)d_in[5];
    const float* b_emb  = (const float*)d_in[6];
    const float* W_st   = (const float*)d_in[7];
    const float* b_st   = (const float*)d_in[8];
    const float* ln_g   = (const float*)d_in[9];
    const float* ln_b   = (const float*)d_in[10];
    const float* lin_W  = (const float*)d_in[11];
    const float* mW1    = (const float*)d_in[12];
    const float* mb1    = (const float*)d_in[13];
    const float* mW2    = (const float*)d_in[14];
    const float* mb2    = (const float*)d_in[15];
    const float* conv_b = (const float*)d_in[16];
    const float* W_dec  = (const float*)d_in[17];
    const float* b_dec  = (const float*)d_in[18];
    float* outp = (float*)d_out;

    // workspace layout (floats)
    float* ws   = (float*)d_ws;
    float* h    = ws;                               // 12.8M  (also reused as conv output)
    float* xl   = h   + (size_t)N_NODES * HID;      // 12.8M
    float* w_e  = xl  + (size_t)N_NODES * HID;      // 1.6M
    float* deg  = w_e + N_EDGES;                    // 100K
    float* dinv = deg + N_NODES;                    // 100K
    _Float16* W16 = (_Float16*)(dinv + N_NODES);    // 16384 halfs (32B-aligned offset)

    const int* row = eidx;
    const int* col = eidx + N_EDGES;

    k_encode<<<N_NODES, HID, 0, stream>>>(x, t, s, W_emb, b_emb, W_st, b_st, ln_g, ln_b, h);

    for (int i = 0; i < N_LAYERS; ++i) {
        k_fill<<<(N_NODES + 255) / 256, 256, 0, stream>>>(deg, 1.0f, N_NODES);
        k_edge_w<<<(N_EDGES + 255) / 256, 256, 0, stream>>>(
            ea, col, mW1 + (size_t)i * EDGE_DIM * HC, mb1 + (size_t)i * HC,
            mW2 + (size_t)i * HC, mb2 + i, w_e, deg);
        k_dinv<<<(N_NODES + 255) / 256, 256, 0, stream>>>(deg, dinv, N_NODES);
        k_wcvt<<<(HID * HID + 255) / 256, 256, 0, stream>>>(lin_W + (size_t)i * HID * HID, W16);
        k_gemm_wmma<<<N_NODES / 16, 256, 0, stream>>>(h, W16, xl);
        k_out_init<<<(N_NODES * HID + 255) / 256, 256, 0, stream>>>(xl, dinv, conv_b + (size_t)i * HID, h);
        k_scatter<<<(N_EDGES * 32) / 256, 256, 0, stream>>>(xl, row, col, w_e, dinv, h);
    }

    k_dec<<<(N_NODES * 32 + 255) / 256, 256, 0, stream>>>(h, W_dec, b_dec, outp);
}
